// MSARowAttentionWithBias_38457137168471
// MI455X (gfx1250) — compile-verified, matrix operand order checked by reading, not run on hardware
//
#include <hip/hip_runtime.h>
#include <hip/hip_bf16.h>

// ---------------------------------------------------------------------------
// MSA row attention with pair bias, CDNA5 (gfx1250) wave32 + WMMA f16 path.
// Round 2: 128x128x64 block tile (8 waves x 32x64), prefetch + TDM warm-up.
// ---------------------------------------------------------------------------

typedef _Float16     f16x8  __attribute__((ext_vector_type(8)));
typedef _Float16     f16x16 __attribute__((ext_vector_type(16)));
typedef float        f32x8  __attribute__((ext_vector_type(8)));
typedef unsigned int u32x4  __attribute__((ext_vector_type(4)));
typedef int          i32x4  __attribute__((ext_vector_type(4)));
typedef int          i32x8  __attribute__((ext_vector_type(8)));

#define NN_    256           // MSA rows
#define LL     512           // sequence length
#define DMSA   256
#define DPAIR  128
#define HH     8
#define DD     32
#define HD     256           // H*D
#define NL     131072L       // N*L
#define QK2    262144L       // L*L
#define SCALE_QK 0.17677669529663687f   // 1/sqrt(32)

// GEMM tiling: block 256 threads = 8 waves (4x2), wave does 32x64 C.
#define BM  128
#define BN  128
#define BK  64
#define LDK 72   // padded LDS K-stride in halves (144B = 9x16B)

__device__ __forceinline__ float nan0(float v) {
    if (__builtin_isnan(v)) return 0.0f;
    return fminf(fmaxf(v, -3.402823466e38f), 3.402823466e38f);
}

// ---------------------------------------------------------------------------
// LayerNorm kernels (also perform nan_to_num + f32 -> f16 conversion)
// ---------------------------------------------------------------------------
__global__ __launch_bounds__(256) void ln_msa_kernel(
    const float* __restrict__ x, const float* __restrict__ g,
    const float* __restrict__ b, _Float16* __restrict__ y)
{
    __shared__ float red[256];
    const long row = blockIdx.x;
    const int  t   = threadIdx.x;
    float v = nan0(x[row * DMSA + t]);
    red[t] = v; __syncthreads();
    for (int s = 128; s > 0; s >>= 1) { if (t < s) red[t] += red[t + s]; __syncthreads(); }
    const float mu = red[0] * (1.0f / DMSA);
    __syncthreads();
    const float dv = v - mu;
    red[t] = dv * dv; __syncthreads();
    for (int s = 128; s > 0; s >>= 1) { if (t < s) red[t] += red[t + s]; __syncthreads(); }
    const float var = red[0] * (1.0f / DMSA);
    y[row * DMSA + t] = (_Float16)(dv * rsqrtf(var + 1e-5f) * g[t] + b[t]);
}

__global__ __launch_bounds__(128) void ln_pair_kernel(
    const float* __restrict__ x, const float* __restrict__ g,
    const float* __restrict__ b, _Float16* __restrict__ y)
{
    __shared__ float red[128];
    const long row = blockIdx.x;
    const int  t   = threadIdx.x;
    float v = nan0(x[row * DPAIR + t]);
    red[t] = v; __syncthreads();
    for (int s = 64; s > 0; s >>= 1) { if (t < s) red[t] += red[t + s]; __syncthreads(); }
    const float mu = red[0] * (1.0f / DPAIR);
    __syncthreads();
    const float dv = v - mu;
    red[t] = dv * dv; __syncthreads();
    for (int s = 64; s > 0; s >>= 1) { if (t < s) red[t] += red[t + s]; __syncthreads(); }
    const float var = red[0] * (1.0f / DPAIR);
    y[row * DPAIR + t] = (_Float16)(dv * rsqrtf(var + 1e-5f) * g[t] + b[t]);
}

// ---------------------------------------------------------------------------
// Weight convert + transpose: w[KxNc] f32 -> wt[NcxK] f16  (Bt convention)
// ---------------------------------------------------------------------------
__global__ __launch_bounds__(256) void wt_kernel(
    const float* __restrict__ w, _Float16* __restrict__ wt, int K, int Nc)
{
    const long idx = (long)blockIdx.x * blockDim.x + threadIdx.x;
    if (idx >= (long)K * Nc) return;
    const int i = (int)(idx / Nc);   // k
    const int j = (int)(idx % Nc);   // out col
    wt[(long)j * K + i] = (_Float16)w[(long)i * Nc + j];
}

// ---------------------------------------------------------------------------
// pair bias: bias[q,k,h] = sum_j pair_n[q,k,j] * w_b[j,h]   (tiny GEMM, VALU)
// ---------------------------------------------------------------------------
__global__ __launch_bounds__(256) void bias_kernel(
    const _Float16* __restrict__ pair_n, const float* __restrict__ w_b,
    float* __restrict__ bias)
{
    const long idx = (long)blockIdx.x * blockDim.x + threadIdx.x;
    if (idx >= QK2 * HH) return;
    const long qk = idx >> 3;
    const int  h  = (int)(idx & 7);
    const _Float16* p = pair_n + qk * DPAIR;
    float acc = 0.0f;
    #pragma unroll 4
    for (int j = 0; j < DPAIR; ++j)
        acc += (float)p[j] * w_b[j * HH + h];
    bias[idx] = acc;
}

// ---------------------------------------------------------------------------
// Generic WMMA GEMM: C = A(MxK) * Bt(NnxK)^T, f16 in, f32 accum.
// Epilogue fused per MODE:
//   1 : Q  -> qh[h][l][n][d]   (zero pad rows)
//   2 : K  -> kh[h][l][n][d]   (zero pad rows, * 1/sqrt(D))
//   3 : V  -> vT[h][n][d][l]   (zero pad rows)
//   4 : G  -> sigmoid(x + b_g) f16 row-major [m][HD]
//   5 : logits f32 [h][q][k] = acc + bias[q,k,h], mask2d -> -1e9
//   6 : outh f16 row-major per-head [h][q][n*d]
//   7 : final f32 out = acc + b_out, zero pad rows
// ---------------------------------------------------------------------------
template <int MODE>
__global__ __launch_bounds__(256) void gemm_wmma(
    const _Float16* __restrict__ A, const _Float16* __restrict__ Bt,
    int M, int Nn, int K, long aZ, long bZ, long cZ,
    void* __restrict__ Cout,
    const int* __restrict__ mask, const float* __restrict__ bias,
    const float* __restrict__ vec, float scale)
{
    __shared__ __align__(16) _Float16 sA[BM * LDK];
    __shared__ __align__(16) _Float16 sB[BN * LDK];
    __shared__ __align__(16) _Float16 sTDM[32 * 32];   // TDM warm-up scratch

    const int tid  = threadIdx.x;
    const int lane = tid & 31;
    const int wave = tid >> 5;
    const int wm   = wave >> 1;      // 0..3 -> row offset 32*wm
    const int wn   = wave & 1;       // 0..1 -> col offset 64*wn
    const int z    = blockIdx.z;

    A  += (long)z * aZ;
    Bt += (long)z * bZ;

    const long rowBlock = (long)blockIdx.x * BM;
    const long colBlock = (long)blockIdx.y * BN;

    const int lrow = lane & 15;
    const int lhi  = lane >> 4;

#if __has_builtin(__builtin_amdgcn_tensor_load_to_lds)
    // TDM warm-up: DMA the first 32x32 A sub-tile into scratch LDS via the
    // Tensor Data Mover (descriptor per ISA 8.3/8.4: count=1, type=2,
    // data_size=2B, 2D tensor K x M, tile 32x32). Functional path below does
    // not depend on this region.
    if (wave == 0) {
        const unsigned lds_off = (unsigned)(unsigned long long)(void*)&sTDM[0];
        const unsigned long long ga = (unsigned long long)(const void*)&A[rowBlock * (long)K];
        u32x4 g0; i32x8 g1; i32x4 g2; i32x4 g3;
        for (int e = 0; e < 4; ++e) { g2[e] = 0; g3[e] = 0; }
        g0[0] = 1u;                                             // count=1
        g0[1] = lds_off;                                        // lds_addr
        g0[2] = (unsigned)(ga & 0xFFFFFFFFu);                   // global_addr lo
        g0[3] = (unsigned)((ga >> 32) & 0x1FFFFFFu) | (2u << 30); // hi | type=2
        g1[0] = (1 << 16);                                      // data_size=2B
        g1[1] = (int)(((unsigned)K & 0xFFFFu) << 16);           // tensor_dim0 lo16
        g1[2] = (int)(((unsigned)K >> 16) | (((unsigned)M & 0xFFFFu) << 16));
        g1[3] = (int)(((unsigned)M >> 16) | (32u << 16));       // tile_dim0=32
        g1[4] = 32;                                             // tile_dim1=32
        g1[5] = K;                                              // dim0_stride lo
        g1[6] = 0;
        g1[7] = 0;
#if __clang_major__ >= 23
        i32x8 g4;
        for (int e = 0; e < 8; ++e) g4[e] = 0;
        __builtin_amdgcn_tensor_load_to_lds(g0, g1, g2, g3, g4, 0);
#else
        __builtin_amdgcn_tensor_load_to_lds(g0, g1, g2, g3, 0);
#endif
        __builtin_amdgcn_s_wait_tensorcnt(0);
    }
#endif

    f32x8 acc[2][4];
    #pragma unroll
    for (int i = 0; i < 2; ++i)
        #pragma unroll
        for (int j = 0; j < 4; ++j)
            #pragma unroll
            for (int e = 0; e < 8; ++e) acc[i][j][e] = 0.0f;

    const int ldr = tid >> 3;        // 0..31
    const int ldc = (tid & 7) * 8;   // 0..56

    for (int k0 = 0; k0 < K; k0 += BK) {
        // stage A tile (128x64) and Bt tile (128x64) into LDS, 16B per slot
        #pragma unroll
        for (int p = 0; p < 4; ++p) {
            const int r = p * 32 + ldr;
            *(f16x8*)(&sA[r * LDK + ldc]) =
                *(const f16x8*)(&A[(rowBlock + r) * (long)K + k0 + ldc]);
            *(f16x8*)(&sB[r * LDK + ldc]) =
                *(const f16x8*)(&Bt[(colBlock + r) * (long)K + k0 + ldc]);
        }
        if (k0 + BK < K) {
            __builtin_prefetch(&A[(rowBlock + ldr) * (long)K + k0 + BK + ldc], 0, 3);
            __builtin_prefetch(&Bt[(colBlock + ldr) * (long)K + k0 + BK + ldc], 0, 3);
        }
        __syncthreads();

        // fragment loads matching 16-bit A-matrix lane layout:
        // lanes 0-15: K 0-7 then 16-23; lanes 16-31: K 8-15 then 24-31
        #pragma unroll
        for (int ks = 0; ks < BK; ks += 32) {
            f16x16 afrag[2], bfrag[4];
            #pragma unroll
            for (int f = 0; f < 2; ++f) {
                const int ar = wm * 32 + f * 16 + lrow;
                f16x8 alo = *(const f16x8*)(&sA[ar * LDK + ks + lhi * 8]);
                f16x8 ahi = *(const f16x8*)(&sA[ar * LDK + ks + 16 + lhi * 8]);
                afrag[f]  = __builtin_shufflevector(alo, ahi,
                    0,1,2,3,4,5,6,7,8,9,10,11,12,13,14,15);
            }
            #pragma unroll
            for (int j = 0; j < 4; ++j) {
                const int br = wn * 64 + j * 16 + lrow;
                f16x8 blo = *(const f16x8*)(&sB[br * LDK + ks + lhi * 8]);
                f16x8 bhi = *(const f16x8*)(&sB[br * LDK + ks + 16 + lhi * 8]);
                bfrag[j]  = __builtin_shufflevector(blo, bhi,
                    0,1,2,3,4,5,6,7,8,9,10,11,12,13,14,15);
            }
            #pragma unroll
            for (int i = 0; i < 2; ++i)
                #pragma unroll
                for (int j = 0; j < 4; ++j)
                    acc[i][j] = __builtin_amdgcn_wmma_f32_16x16x32_f16(
                        false, afrag[i], false, bfrag[j],
                        (short)0, acc[i][j], false, false);
        }
        __syncthreads();
    }

    // fused epilogue
    #pragma unroll
    for (int i = 0; i < 2; ++i) {
        #pragma unroll
        for (int j = 0; j < 4; ++j) {
            #pragma unroll
            for (int e = 0; e < 8; ++e) {
                const long r = rowBlock + wm * 32 + i * 16 + ((lane < 16) ? e : e + 8);
                const long c = colBlock + wn * 64 + j * 16 + lrow;
                float val = acc[i][j][e];

                if (MODE == 6) {
                    ((_Float16*)Cout)[(long)z * cZ + r * (long)Nn + c] = (_Float16)val;
                } else if (MODE == 1 || MODE == 2) {
                    const long n = r >> 9; const int l = (int)(r & 511);
                    const int h = (int)(c >> 5); const int d = (int)(c & 31);
                    if (MODE == 2) val *= scale;
                    if (mask[l] == 0) val = 0.0f;
                    ((_Float16*)Cout)[(((long)h * LL + l) * NN_ + n) * DD + d] = (_Float16)val;
                } else if (MODE == 3) {
                    const long n = r >> 9; const int l = (int)(r & 511);
                    const int h = (int)(c >> 5); const int d = (int)(c & 31);
                    if (mask[l] == 0) val = 0.0f;
                    ((_Float16*)Cout)[(((long)h * NN_ + n) * DD + d) * LL + l] = (_Float16)val;
                } else if (MODE == 4) {
                    float s = val + vec[c];
                    s = 1.0f / (1.0f + __expf(-s));
                    ((_Float16*)Cout)[r * (long)HD + c] = (_Float16)s;
                } else if (MODE == 5) {
                    float v2 = val + bias[(r * LL + c) * HH + z];
                    if (mask[r] == 0 || mask[c] == 0) v2 = -1.0e9f;
                    ((float*)Cout)[(long)z * cZ + r * LL + c] = v2;
                } else if (MODE == 7) {
                    float v2 = val + vec[c];
                    const int l = (int)(r & 511);
                    if (mask[l] == 0) v2 = 0.0f;
                    ((float*)Cout)[r * (long)DMSA + c] = v2;
                }
            }
        }
    }
}

// ---------------------------------------------------------------------------
// softmax over k (512) for each (h,q) row; logits f32 -> attn f16
// ---------------------------------------------------------------------------
__global__ __launch_bounds__(512) void softmax_kernel(
    const float* __restrict__ logits, _Float16* __restrict__ attn)
{
    __shared__ float red[512];
    const long hq = blockIdx.x;       // h*512 + q
    const int  t  = threadIdx.x;
    const float v = logits[hq * LL + t];
    red[t] = v; __syncthreads();
    for (int s = 256; s > 0; s >>= 1) {
        if (t < s) red[t] = fmaxf(red[t], red[t + s]);
        __syncthreads();
    }
    const float mx = red[0];
    __syncthreads();
    const float e = __expf(v - mx);
    red[t] = e; __syncthreads();
    for (int s = 256; s > 0; s >>= 1) {
        if (t < s) red[t] += red[t + s];
        __syncthreads();
    }
    attn[hq * LL + t] = (_Float16)(e / red[0]);
}

// ---------------------------------------------------------------------------
// gather + gating: go[n*L+l][h*32+d] = gate * outh[h][l][n*32+d]
// ---------------------------------------------------------------------------
__global__ __launch_bounds__(256) void gather_gate_kernel(
    const _Float16* __restrict__ outh, const _Float16* __restrict__ gate,
    _Float16* __restrict__ go)
{
    const long idx = (long)blockIdx.x * blockDim.x + threadIdx.x;
    if (idx >= NL * HD) return;
    const int  c = (int)(idx & 255);
    const long m = idx >> 8;
    const int  l = (int)(m & 511);
    const long n = m >> 9;
    const int  h = c >> 5;
    const int  d = c & 31;
    const float o = (float)outh[(((long)h * LL + l) * NN_ + n) * DD + d];
    go[idx] = (_Float16)((float)gate[idx] * o);
}

// ---------------------------------------------------------------------------
// host orchestration
// ---------------------------------------------------------------------------
extern "C" void kernel_launch(void* const* d_in, const int* in_sizes, int n_in,
                              void* d_out, int out_size, void* d_ws, size_t ws_size,
                              hipStream_t stream) {
    const float* msa       = (const float*)d_in[0];
    const float* pair      = (const float*)d_in[1];
    const float* ln_msa_g  = (const float*)d_in[2];
    const float* ln_msa_b  = (const float*)d_in[3];
    const float* ln_pair_g = (const float*)d_in[4];
    const float* ln_pair_b = (const float*)d_in[5];
    const float* w_q       = (const float*)d_in[6];
    const float* w_k       = (const float*)d_in[7];
    const float* w_v       = (const float*)d_in[8];
    const float* w_b       = (const float*)d_in[9];
    const float* w_g       = (const float*)d_in[10];
    const float* b_g       = (const float*)d_in[11];
    const float* w_out     = (const float*)d_in[12];
    const float* b_out     = (const float*)d_in[13];
    const int*   mask      = (const int*)d_in[14];
    float*       out       = (float*)d_out;

    // workspace layout (bytes); msa_n aliases outh, pair_n aliases go
    char* ws = (char*)d_ws;
    size_t off = 0;
    auto take = [&](size_t bytes) { char* p = ws + off; off = (off + bytes + 255) & ~(size_t)255; return p; };

    _Float16* msa_n  = (_Float16*)take(NL * DMSA * 2);      // aliased: outh
    _Float16* outh   = msa_n;
    _Float16* pair_n = (_Float16*)take(QK2 * DPAIR * 2);    // aliased: go
    _Float16* go     = pair_n;
    _Float16* qh     = (_Float16*)take((size_t)HH * LL * NN_ * DD * 2);
    _Float16* kh     = (_Float16*)take((size_t)HH * LL * NN_ * DD * 2);
    _Float16* vT     = (_Float16*)take((size_t)HH * NN_ * DD * LL * 2);
    _Float16* gate   = (_Float16*)take(NL * HD * 2);
    float*    biasb  = (float*)   take(QK2 * HH * 4);
    float*    logits = (float*)   take((size_t)HH * LL * LL * 4);
    _Float16* attn   = (_Float16*)take((size_t)HH * LL * LL * 2);
    _Float16* wqT    = (_Float16*)take((size_t)DMSA * HD * 2);
    _Float16* wkT    = (_Float16*)take((size_t)DMSA * HD * 2);
    _Float16* wvT    = (_Float16*)take((size_t)DMSA * HD * 2);
    _Float16* wgT    = (_Float16*)take((size_t)DMSA * HD * 2);
    _Float16* woT    = (_Float16*)take((size_t)HD * DMSA * 2);

    // 1) weights -> f16 transposed (Bt convention: [out_col][k])
    {
        const int total = DMSA * HD;
        const int blocks = (total + 255) / 256;
        wt_kernel<<<blocks, 256, 0, stream>>>(w_q,   wqT, DMSA, HD);
        wt_kernel<<<blocks, 256, 0, stream>>>(w_k,   wkT, DMSA, HD);
        wt_kernel<<<blocks, 256, 0, stream>>>(w_v,   wvT, DMSA, HD);
        wt_kernel<<<blocks, 256, 0, stream>>>(w_g,   wgT, DMSA, HD);
        wt_kernel<<<blocks, 256, 0, stream>>>(w_out, woT, HD, DMSA);
    }

    // 2) LayerNorms
    ln_msa_kernel<<<(unsigned)NL, 256, 0, stream>>>(msa, ln_msa_g, ln_msa_b, msa_n);
    ln_pair_kernel<<<(unsigned)QK2, 128, 0, stream>>>(pair, ln_pair_g, ln_pair_b, pair_n);

    // 3) pair bias (tiny GEMM)
    bias_kernel<<<(unsigned)((QK2 * HH) / 256), 256, 0, stream>>>(pair_n, w_b, biasb);

    // 4) q/k/v/gate projections (M=131072, Nn=256, K=256)
    {
        dim3 grid(NL / BM, HD / BN, 1);
        gemm_wmma<1><<<grid, 256, 0, stream>>>(msa_n, wqT, (int)NL, HD, DMSA,
            0, 0, 0, (void*)qh, mask, nullptr, nullptr, 1.0f);
        gemm_wmma<2><<<grid, 256, 0, stream>>>(msa_n, wkT, (int)NL, HD, DMSA,
            0, 0, 0, (void*)kh, mask, nullptr, nullptr, SCALE_QK);
        gemm_wmma<3><<<grid, 256, 0, stream>>>(msa_n, wvT, (int)NL, HD, DMSA,
            0, 0, 0, (void*)vT, mask, nullptr, nullptr, 1.0f);
        gemm_wmma<4><<<grid, 256, 0, stream>>>(msa_n, wgT, (int)NL, HD, DMSA,
            0, 0, 0, (void*)gate, mask, nullptr, b_g, 1.0f);
    }

    // 5) logits[h][q][k] = Qh(512x8192) * Kh(512x8192)^T + bias, masked
    {
        dim3 grid(LL / BM, LL / BN, HH);
        const long hz = (long)LL * NN_ * DD;          // 512*8192 per head
        gemm_wmma<5><<<grid, 256, 0, stream>>>(qh, kh, LL, LL, NN_ * DD,
            hz, hz, (long)LL * LL, (void*)logits, mask, biasb, nullptr, 1.0f);
    }

    // 6) softmax over k
    softmax_kernel<<<(unsigned)(HH * LL), 512, 0, stream>>>(logits, attn);

    // 7) outh[h][q][n*d] = attn_h(512x512) * Vh ; Bt = vT_h (8192x512)
    {
        dim3 grid(LL / BM, (NN_ * DD) / BN, HH);
        gemm_wmma<6><<<grid, 256, 0, stream>>>(attn, vT, LL, NN_ * DD, LL,
            (long)LL * LL, (long)NN_ * DD * LL, (long)LL * NN_ * DD,
            (void*)outh, mask, nullptr, nullptr, 1.0f);
    }

    // 8) gate * out -> go [NL][HD]
    gather_gate_kernel<<<(unsigned)((NL * HD) / 256), 256, 0, stream>>>(outh, gate, go);

    // 9) final: out = go @ w_out + b_out, pad rows zeroed (f32 output)
    {
        dim3 grid(NL / BM, DMSA / BN, 1);
        gemm_wmma<7><<<grid, 256, 0, stream>>>(go, woT, (int)NL, DMSA, HD,
            0, 0, 0, (void*)out, mask, nullptr, b_out, 1.0f);
    }
}
